// SparseEGNNFlowMatching_11759620456696
// MI455X (gfx1250) — compile-verified, matrix operand order checked by reading, not run on hardware
//
#include <hip/hip_runtime.h>
#include <hip/hip_bf16.h>

// ---------------- problem constants ----------------
#define BB    4
#define NN    2500
#define BN    10000
#define NE    160000
#define HH    128
#define DD    6
#define KK    16

// ---------------- WMMA plumbing ----------------
typedef __bf16        v16bf __attribute__((ext_vector_type(16)));
typedef float         v8f   __attribute__((ext_vector_type(8)));
typedef unsigned int  u32x4 __attribute__((ext_vector_type(4)));

union BFPack { u32x4 q[2]; v16bf v; };

__device__ __forceinline__ unsigned short f32_to_bf16(float f) {
    unsigned int u = __float_as_uint(f);
    u += 0x7FFFu + ((u >> 16) & 1u);            // round-to-nearest-even
    return (unsigned short)(u >> 16);
}

__device__ __forceinline__ float silu_f(float v) { return v / (1.f + __expf(-v)); }

// A operand: LDS tile, lane = M row (lane&15), K halves split on lane>=16.
// B operand: N-major packed bf16 weights, lane = column, same K split.
__device__ __forceinline__ v8f wmma_bf16(const unsigned short* a,
                                         const unsigned short* b, v8f c) {
    BFPack pa, pb;
    pa.q[0] = *(const u32x4*)(a);
    pa.q[1] = *(const u32x4*)(a + 16);
    pb.q[0] = *(const u32x4*)(b);
    pb.q[1] = *(const u32x4*)(b + 16);
    return __builtin_amdgcn_wmma_f32_16x16x32_bf16(
        false, pa.v, false, pb.v, (short)0, c, false, false);
}

__device__ __forceinline__ void atomic_add_f32(float* p, float v) {
    __hip_atomic_fetch_add(p, v, __ATOMIC_RELAXED, __HIP_MEMORY_SCOPE_AGENT);
}

// ---------------- small setup kernels ----------------

// gamma/beta from t (B=4): t_emb = silu(t*te_w1+te_b1)@te_w2+te_b2 ; gb = t_emb@film_w+film_b
__global__ void __launch_bounds__(128) film_params_kernel(
    const float* __restrict__ t, const float* __restrict__ te_w1,
    const float* __restrict__ te_b1, const float* __restrict__ te_w2,
    const float* __restrict__ te_b2, const float* __restrict__ film_w,
    const float* __restrict__ film_b, float* __restrict__ gamma,
    float* __restrict__ beta) {
    __shared__ float s1[BB][HH];
    __shared__ float s2[BB][HH];
    const int j = threadIdx.x;
    for (int b = 0; b < BB; ++b)
        s1[b][j] = silu_f(t[b] * te_w1[j] + te_b1[j]);
    __syncthreads();
    for (int b = 0; b < BB; ++b) {
        float acc = te_b2[j];
        for (int k = 0; k < HH; ++k) acc += s1[b][k] * te_w2[k * HH + j];
        s2[b][j] = acc;
    }
    __syncthreads();
    for (int b = 0; b < BB; ++b) {
        float g = film_b[j], be = film_b[j + HH];
        for (int k = 0; k < HH; ++k) {
            float tv = s2[b][k];
            g  += tv * film_w[k * 2 * HH + j];
            be += tv * film_w[k * 2 * HH + j + HH];
        }
        gamma[b * HH + j] = g;
        beta [b * HH + j] = be;
    }
}

// rij / dij^2 per edge
__global__ void __launch_bounds__(256) edge_geom_kernel(
    const float* __restrict__ x, const long long* __restrict__ eidx,
    float* __restrict__ rij, float* __restrict__ dsq) {
    int e = blockIdx.x * 256 + threadIdx.x;
    if (e >= NE) return;
    int r = (int)eidx[e];
    int c = (int)eidx[NE + e];
    float rx = x[r * 2]     - x[c * 2];
    float ry = x[r * 2 + 1] - x[c * 2 + 1];
    rij[e * 2]     = rx;
    rij[e * 2 + 1] = ry;
    dsq[e] = rx * rx + ry * ry;
}

// h = broadcast(node_emb); vacc = 0
__global__ void __launch_bounds__(256) init_h_kernel(
    const float* __restrict__ node_emb, float* __restrict__ h,
    float* __restrict__ vacc) {
    int i = blockIdx.x * 256 + threadIdx.x;
    if (i < BN * HH) h[i] = node_emb[i & (HH - 1)];
    if (i < BN * 2)  vacc[i] = 0.f;
}

// hf = h*(1+gamma)+beta ; agg = 0
__global__ void __launch_bounds__(256) film_apply_kernel(
    const float* __restrict__ h, const float* __restrict__ gamma,
    const float* __restrict__ beta, float* __restrict__ hf,
    float* __restrict__ agg) {
    int i = blockIdx.x * 256 + threadIdx.x;
    if (i >= BN * HH) return;
    int f = i & (HH - 1);
    int node = i >> 7;
    int b = node / NN;
    hf[i]  = h[i] * (1.f + gamma[b * HH + f]) + beta[b * HH + f];
    agg[i] = 0.f;
}

// Transpose + bf16-pack a (Kuse x 128) f32 weight into N-major [128][Kuse] bf16
__global__ void __launch_bounds__(256) pack_wt_kernel(
    const float* __restrict__ src, unsigned short* __restrict__ dst, int Kuse) {
    int i = blockIdx.x * 256 + threadIdx.x;
    if (i >= HH * Kuse) return;
    int n = i / Kuse, k = i % Kuse;
    dst[n * Kuse + k] = f32_to_bf16(src[k * HH + n]);
}

// ---------------- WMMA edge MLP ----------------
// One block = 16 edges. 8 waves, each owns a 16-column N-tile.
__global__ void __launch_bounds__(256) edge_mlp_kernel(
    const float* __restrict__ hf, const long long* __restrict__ eidx,
    const float* __restrict__ dsq,
    const unsigned short* __restrict__ w1t, const float* __restrict__ w1last,
    const float* __restrict__ b1,
    const unsigned short* __restrict__ w2t, const float* __restrict__ b2,
    float* __restrict__ agg) {
    __shared__ unsigned short sA [16 * 264];   // e_in tile, bf16, K=0..255
    __shared__ unsigned short sA2[16 * 136];   // silu(layer1) tile, bf16
    __shared__ float sDij[16];
    __shared__ int   sRow[16], sCol[16];

    const int tid = threadIdx.x;
    const int e0  = blockIdx.x * 16;
    if (tid < 16) {
        sRow[tid] = (int)eidx[e0 + tid];
        sDij[tid] = dsq[e0 + tid];
    } else if (tid < 32) {
        sCol[tid - 16] = (int)eidx[NE + e0 + tid - 16];
    }
    __syncthreads();
    for (int idx = tid; idx < 16 * 256; idx += 256) {   // gather + bf16 convert
        int m = idx >> 8, f = idx & 255;
        float v = (f < HH) ? hf[sRow[m] * HH + f] : hf[sCol[m] * HH + (f - HH)];
        sA[m * 264 + f] = f32_to_bf16(v);
    }
    __syncthreads();

    const int lane  = tid & 31;
    const int wv    = tid >> 5;
    const int nl    = lane & 15;
    const int n     = wv * 16 + nl;
    const int khalf = (lane >> 4) << 3;          // 0 or 8
    const int mbase = (lane < 16) ? 0 : 8;

    v8f acc = {0.f, 0.f, 0.f, 0.f, 0.f, 0.f, 0.f, 0.f};
#pragma unroll
    for (int kb = 0; kb < 8; ++kb)
        acc = wmma_bf16(&sA[nl * 264 + kb * 32 + khalf],
                        &w1t[n * 256 + kb * 32 + khalf], acc);

    const float wl = w1last[n];   // dij^2 column of ew1 (K=256), f32
    const float bv = b1[n];
#pragma unroll
    for (int r = 0; r < 8; ++r) {
        float v = acc[r] + sDij[mbase + r] * wl + bv;
        sA2[(mbase + r) * 136 + n] = f32_to_bf16(silu_f(v));
    }
    __syncthreads();

    v8f acc2 = {0.f, 0.f, 0.f, 0.f, 0.f, 0.f, 0.f, 0.f};
#pragma unroll
    for (int kb = 0; kb < 4; ++kb)
        acc2 = wmma_bf16(&sA2[nl * 136 + kb * 32 + khalf],
                         &w2t[n * 128 + kb * 32 + khalf], acc2);
    const float b2n = b2[n];
#pragma unroll
    for (int r = 0; r < 8; ++r) {
        float v = silu_f(acc2[r] + b2n);                // m_ij
        atomic_add_f32(&agg[sRow[mbase + r] * HH + n], v);  // segment_sum
    }
}

// ---------------- WMMA node MLP + residual + LayerNorm ----------------
__global__ void __launch_bounds__(256) node_mlp_kernel(
    const float* __restrict__ hf, const float* __restrict__ agg,
    const unsigned short* __restrict__ w1t, const float* __restrict__ b1,
    const unsigned short* __restrict__ w2t, const float* __restrict__ b2,
    const float* __restrict__ lng, const float* __restrict__ lnb,
    float* __restrict__ h) {
    __shared__ unsigned short sA [16 * 264];
    __shared__ unsigned short sA2[16 * 136];
    __shared__ float sLN[16 * 132];
    __shared__ float sMean[16], sRstd[16];

    const int tid = threadIdx.x;
    const int n0  = blockIdx.x * 16;
    for (int idx = tid; idx < 16 * 256; idx += 256) {
        int m = idx >> 8, f = idx & 255;
        int node = n0 + m;
        float v = (f < HH) ? hf[node * HH + f]
                           : agg[node * HH + (f - HH)] * (1.f / (float)KK);
        sA[m * 264 + f] = f32_to_bf16(v);
    }
    __syncthreads();

    const int lane  = tid & 31;
    const int wv    = tid >> 5;
    const int nl    = lane & 15;
    const int n     = wv * 16 + nl;
    const int khalf = (lane >> 4) << 3;
    const int mbase = (lane < 16) ? 0 : 8;

    v8f acc = {0.f, 0.f, 0.f, 0.f, 0.f, 0.f, 0.f, 0.f};
#pragma unroll
    for (int kb = 0; kb < 8; ++kb)
        acc = wmma_bf16(&sA[nl * 264 + kb * 32 + khalf],
                        &w1t[n * 256 + kb * 32 + khalf], acc);
    const float bv = b1[n];
#pragma unroll
    for (int r = 0; r < 8; ++r)
        sA2[(mbase + r) * 136 + n] = f32_to_bf16(silu_f(acc[r] + bv));
    __syncthreads();

    v8f acc2 = {0.f, 0.f, 0.f, 0.f, 0.f, 0.f, 0.f, 0.f};
#pragma unroll
    for (int kb = 0; kb < 4; ++kb)
        acc2 = wmma_bf16(&sA2[nl * 136 + kb * 32 + khalf],
                         &w2t[n * 128 + kb * 32 + khalf], acc2);
    const float b2n = b2[n];
#pragma unroll
    for (int r = 0; r < 8; ++r) {
        int m = mbase + r;
        sLN[m * 132 + n] = hf[(n0 + m) * HH + n] + (acc2[r] + b2n);  // residual
    }
    __syncthreads();

    if (tid < 16) {                                   // per-row LN stats
        float s = 0.f, s2 = 0.f;
        for (int f = 0; f < HH; ++f) {
            float v = sLN[tid * 132 + f];
            s += v; s2 += v * v;
        }
        float mean = s * (1.f / (float)HH);
        float var  = s2 * (1.f / (float)HH) - mean * mean;
        sMean[tid] = mean;
        sRstd[tid] = rsqrtf(var + 1e-5f);
    }
    __syncthreads();
    for (int idx = tid; idx < 16 * HH; idx += 256) {
        int m = idx >> 7, f = idx & (HH - 1);
        float v = (sLN[m * 132 + f] - sMean[m]) * sRstd[m];
        h[(n0 + m) * HH + f] = v * lng[f] + lnb[f];
    }
}

// ---------------- final coordinate head ----------------
__global__ void __launch_bounds__(256) final_edge_kernel(
    const float* __restrict__ h, const long long* __restrict__ eidx,
    const float* __restrict__ dsq, const float* __restrict__ rij,
    const unsigned short* __restrict__ w1t, const float* __restrict__ w1last,
    const float* __restrict__ b1, const float* __restrict__ cw2,
    float* __restrict__ vacc) {
    __shared__ unsigned short sA[16 * 264];
    __shared__ float sHid[16 * 132];
    __shared__ float sDij[16];
    __shared__ int   sRow[16], sCol[16];

    const int tid = threadIdx.x;
    const int e0  = blockIdx.x * 16;
    if (tid < 16) {
        sRow[tid] = (int)eidx[e0 + tid];
        sDij[tid] = dsq[e0 + tid];
    } else if (tid < 32) {
        sCol[tid - 16] = (int)eidx[NE + e0 + tid - 16];
    }
    __syncthreads();
    for (int idx = tid; idx < 16 * 256; idx += 256) {
        int m = idx >> 8, f = idx & 255;
        float v = (f < HH) ? h[sRow[m] * HH + f] : h[sCol[m] * HH + (f - HH)];
        sA[m * 264 + f] = f32_to_bf16(v);
    }
    __syncthreads();

    const int lane  = tid & 31;
    const int wv    = tid >> 5;
    const int nl    = lane & 15;
    const int n     = wv * 16 + nl;
    const int khalf = (lane >> 4) << 3;
    const int mbase = (lane < 16) ? 0 : 8;

    v8f acc = {0.f, 0.f, 0.f, 0.f, 0.f, 0.f, 0.f, 0.f};
#pragma unroll
    for (int kb = 0; kb < 8; ++kb)
        acc = wmma_bf16(&sA[nl * 264 + kb * 32 + khalf],
                        &w1t[n * 256 + kb * 32 + khalf], acc);
    const float wl = w1last[n];
    const float bv = b1[n];
#pragma unroll
    for (int r = 0; r < 8; ++r) {
        int m = mbase + r;
        sHid[m * 132 + n] = silu_f(acc[r] + sDij[m] * wl + bv);
    }
    __syncthreads();

    if (tid < 16) {
        float s = 0.f;
        for (int f = 0; f < HH; ++f) s += sHid[tid * 132 + f] * cw2[f];
        float w   = tanhf(s * (1.f / 10.f));
        float dij = sqrtf(sDij[tid] + 1e-8f);
        float sc  = w / (dij * (float)KK);
        int e     = e0 + tid;
        int node  = sRow[tid];
        atomic_add_f32(&vacc[node * 2],     sc * rij[e * 2]);
        atomic_add_f32(&vacc[node * 2 + 1], sc * rij[e * 2 + 1]);
    }
}

// per-batch mean subtraction
__global__ void __launch_bounds__(256) finalize_kernel(
    const float* __restrict__ vacc, float* __restrict__ out) {
    __shared__ float sx[256], sy[256];
    const int b = blockIdx.x, tid = threadIdx.x;
    float px = 0.f, py = 0.f;
    for (int i = tid; i < NN; i += 256) {
        px += vacc[(b * NN + i) * 2];
        py += vacc[(b * NN + i) * 2 + 1];
    }
    sx[tid] = px; sy[tid] = py;
    __syncthreads();
    for (int s = 128; s > 0; s >>= 1) {
        if (tid < s) { sx[tid] += sx[tid + s]; sy[tid] += sy[tid + s]; }
        __syncthreads();
    }
    float mx = sx[0] * (1.f / (float)NN);
    float my = sy[0] * (1.f / (float)NN);
    for (int i = tid; i < NN; i += 256) {
        out[(b * NN + i) * 2]     = vacc[(b * NN + i) * 2]     - mx;
        out[(b * NN + i) * 2 + 1] = vacc[(b * NN + i) * 2 + 1] - my;
    }
}

// ---------------- host launch ----------------
extern "C" void kernel_launch(void* const* d_in, const int* in_sizes, int n_in,
                              void* d_out, int out_size, void* d_ws, size_t ws_size,
                              hipStream_t stream) {
    (void)in_sizes; (void)n_in; (void)out_size; (void)ws_size;
    const float*     x        = (const float*)d_in[0];
    const float*     t        = (const float*)d_in[1];
    const long long* eidx     = (const long long*)d_in[2];
    const float*     node_emb = (const float*)d_in[3];
    const float*     te_w1    = (const float*)d_in[4];
    const float*     te_b1    = (const float*)d_in[5];
    const float*     te_w2    = (const float*)d_in[6];
    const float*     te_b2    = (const float*)d_in[7];
    const float*     film_w   = (const float*)d_in[8];
    const float*     film_b   = (const float*)d_in[9];
    const float*     ln_g     = (const float*)d_in[10];
    const float*     ln_b     = (const float*)d_in[11];
    const float*     ew1      = (const float*)d_in[12];
    const float*     eb1      = (const float*)d_in[13];
    const float*     ew2      = (const float*)d_in[14];
    const float*     eb2      = (const float*)d_in[15];
    const float*     nw1      = (const float*)d_in[16];
    const float*     nb1      = (const float*)d_in[17];
    const float*     nw2      = (const float*)d_in[18];
    const float*     nb2      = (const float*)d_in[19];
    const float*     cw1      = (const float*)d_in[20];
    const float*     cb1      = (const float*)d_in[21];
    const float*     cw2      = (const float*)d_in[22];

    char* wsb = (char*)d_ws;
    size_t off = 0;
    auto alloc = [&](size_t bytes) -> void* {
        void* p = wsb + off;
        off = (off + bytes + 255) & ~(size_t)255;
        return p;
    };
    float* h     = (float*)alloc((size_t)BN * HH * 4);
    float* hf    = (float*)alloc((size_t)BN * HH * 4);
    float* agg   = (float*)alloc((size_t)BN * HH * 4);
    float* rij   = (float*)alloc((size_t)NE * 2 * 4);
    float* dsq   = (float*)alloc((size_t)NE * 4);
    float* gamma = (float*)alloc((size_t)BB * HH * 4);
    float* beta  = (float*)alloc((size_t)BB * HH * 4);
    float* vacc  = (float*)alloc((size_t)BN * 2 * 4);
    unsigned short* ew1t = (unsigned short*)alloc((size_t)DD * HH * 256 * 2);
    unsigned short* ew2t = (unsigned short*)alloc((size_t)DD * HH * 128 * 2);
    unsigned short* nw1t = (unsigned short*)alloc((size_t)DD * HH * 256 * 2);
    unsigned short* nw2t = (unsigned short*)alloc((size_t)DD * HH * 128 * 2);
    unsigned short* cw1t = (unsigned short*)alloc((size_t)HH * 256 * 2);

    // weight packing (bf16, N-major)
    for (int d = 0; d < DD; ++d) {
        pack_wt_kernel<<<(HH * 256 + 255) / 256, 256, 0, stream>>>(
            ew1 + (size_t)d * 257 * HH, ew1t + (size_t)d * HH * 256, 256);
        pack_wt_kernel<<<(HH * 128 + 255) / 256, 256, 0, stream>>>(
            ew2 + (size_t)d * 128 * HH, ew2t + (size_t)d * HH * 128, 128);
        pack_wt_kernel<<<(HH * 256 + 255) / 256, 256, 0, stream>>>(
            nw1 + (size_t)d * 256 * HH, nw1t + (size_t)d * HH * 256, 256);
        pack_wt_kernel<<<(HH * 128 + 255) / 256, 256, 0, stream>>>(
            nw2 + (size_t)d * 128 * HH, nw2t + (size_t)d * HH * 128, 128);
    }
    pack_wt_kernel<<<(HH * 256 + 255) / 256, 256, 0, stream>>>(cw1, cw1t, 256);

    film_params_kernel<<<1, 128, 0, stream>>>(t, te_w1, te_b1, te_w2, te_b2,
                                              film_w, film_b, gamma, beta);
    edge_geom_kernel<<<NE / 256, 256, 0, stream>>>(x, eidx, rij, dsq);
    init_h_kernel<<<(BN * HH + 255) / 256, 256, 0, stream>>>(node_emb, h, vacc);

    for (int d = 0; d < DD; ++d) {
        film_apply_kernel<<<(BN * HH + 255) / 256, 256, 0, stream>>>(
            h, gamma, beta, hf, agg);
        edge_mlp_kernel<<<NE / 16, 256, 0, stream>>>(
            hf, eidx, dsq,
            ew1t + (size_t)d * HH * 256,
            ew1 + (size_t)d * 257 * HH + 256 * HH,   // dij^2 row, f32
            eb1 + (size_t)d * HH,
            ew2t + (size_t)d * HH * 128,
            eb2 + (size_t)d * HH, agg);
        node_mlp_kernel<<<BN / 16, 256, 0, stream>>>(
            hf, agg,
            nw1t + (size_t)d * HH * 256, nb1 + (size_t)d * HH,
            nw2t + (size_t)d * HH * 128, nb2 + (size_t)d * HH,
            ln_g + (size_t)d * HH, ln_b + (size_t)d * HH, h);
    }

    final_edge_kernel<<<NE / 16, 256, 0, stream>>>(
        h, eidx, dsq, rij, cw1t, cw1 + 256 * HH, cb1, cw2, vacc);
    finalize_kernel<<<BB, 256, 0, stream>>>(vacc, (float*)d_out);
}
